// SparseAttention_2662879724001
// MI455X (gfx1250) — matrix operand order repaired and checked by reading, not compiled, not run on hardware
//
#include <hip/hip_runtime.h>
#include <hip/hip_bf16.h>

typedef __attribute__((ext_vector_type(16))) _Float16 v16h;
typedef __attribute__((ext_vector_type(8)))  _Float16 v8h;
typedef __attribute__((ext_vector_type(2)))  _Float16 v2h;
typedef __attribute__((ext_vector_type(8)))  float    v8f;

#define SEQ    4096
#define DIM    1024
#define NTILE  32      // K/V rows consumed per iteration
#define DCHUNK 128     // head-dim columns owned by each wave (8 waves * 128 = 1024)
#define VTS    40      // Vt column stride in halves (32 used + 8 pad; 80B keeps 16B align)

__device__ __forceinline__ v8f zero8() {
    v8f z;
#pragma unroll
    for (int i = 0; i < 8; ++i) z[i] = 0.0f;
    return z;
}

__global__ __launch_bounds__(256)
void SparseAttention_flash_kernel(const float* __restrict__ q,
                                  const float* __restrict__ k,
                                  const float* __restrict__ v,
                                  float* __restrict__ out)
{
    __shared__ _Float16 Vt[8 * DCHUNK * VTS];   // 80KB: per-wave transposed f16 V tile
    __shared__ float    S_lds[16 * 32];         // cross-wave partial-score reduction
    __shared__ _Float16 P_lds[16 * 32];         // softmaxed probabilities (f16)
    __shared__ float    scale_lds[16];          // per-row online-softmax rescale
    __shared__ float    l_lds[16];              // per-row denominator

    const int tid   = threadIdx.x;
    const int lane  = tid & 31;
    const int wave  = tid >> 5;
    const int b     = blockIdx.x >> 8;          // 4 batches
    const int q0    = (blockIdx.x & 255) << 4;  // 256 strips of 16 query rows
    const int n16   = lane & 15;
    const int hi    = lane >> 4;
    const int chunk = wave * DCHUNK;

    const float* Qb = q + (size_t)b * SEQ * DIM;
    const float* Kb = k + (size_t)b * SEQ * DIM;
    const float* Vb = v + (size_t)b * SEQ * DIM;

    _Float16* vt = &Vt[wave * DCHUNK * VTS];    // this wave's private transpose buffer

    // S_lds must start zeroed for the first tile's atomic reduction
    S_lds[tid]       = 0.0f;
    S_lds[tid + 256] = 0.0f;
    __syncthreads();

    // ---- Load Q strip once as WMMA A-fragments (f32 -> f16 on the fly) ----
    // A fragment as loaded here (k order shared with B): M = lane%16; lanes 0-15: K = 0..15;
    //                       lanes 16-31: K = 16..31.
    const int k0a = hi ? 16 : 0;
    v16h qfrag[4];
    {
        const float* qrow = Qb + (size_t)(q0 + n16) * DIM + chunk;
#pragma unroll
        for (int kt = 0; kt < 4; ++kt) {
            v16h f;
#pragma unroll
            for (int j = 0; j < 8; ++j) {
                f[j]     = (_Float16)qrow[kt * 32 + k0a + j];
                f[j + 8] = (_Float16)qrow[kt * 32 + k0a + 8 + j];
            }
            qfrag[kt] = f;
        }
    }

    // B fragment as loaded here (same k order as A): N = lane%16; lanes 0-15 hold K=0..15, lanes 16-31 K=16..31.
    const int kb0   = hi ? 16 : 0;
    const int rbase = hi ? 8 : 0;   // C/D layout: VGPR i -> row rbase+i, col = lane%16

    v8f o[8];
#pragma unroll
    for (int ct = 0; ct < 8; ++ct) o[ct] = zero8();

    float m_reg = -__builtin_inff();   // online-softmax state (wave 0 only)
    float l_reg = 0.0f;

    for (int s0 = 0; s0 < SEQ; s0 += NTILE) {
        // ---------- stage V tile (own 32x128 chunk) into LDS, transposed, f16 ----------
        // Coalesced: lane reads cols chunk+4*lane of row pairs; stores packed (r,r+1) b32.
        {
            const float* vsrc = Vb + (size_t)s0 * DIM + chunk + lane * 4;
#pragma unroll
            for (int rp = 0; rp < 16; ++rp) {
                float4 r0 = *(const float4*)(vsrc + (size_t)(2 * rp)     * DIM);
                float4 r1 = *(const float4*)(vsrc + (size_t)(2 * rp + 1) * DIM);
                v2h p;
                p[0] = (_Float16)r0.x; p[1] = (_Float16)r1.x;
                *(v2h*)&vt[(lane * 4 + 0) * VTS + 2 * rp] = p;
                p[0] = (_Float16)r0.y; p[1] = (_Float16)r1.y;
                *(v2h*)&vt[(lane * 4 + 1) * VTS + 2 * rp] = p;
                p[0] = (_Float16)r0.z; p[1] = (_Float16)r1.z;
                *(v2h*)&vt[(lane * 4 + 2) * VTS + 2 * rp] = p;
                p[0] = (_Float16)r0.w; p[1] = (_Float16)r1.w;
                *(v2h*)&vt[(lane * 4 + 3) * VTS + 2 * rp] = p;
            }
        }

        // ---------- partial S = Q_chunk * K_chunk^T (per-wave, 128 of 1024 dims) ----------
#pragma unroll
        for (int nt = 0; nt < 2; ++nt) {
            const float* krow = Kb + (size_t)(s0 + nt * 16 + n16) * DIM + chunk + kb0;
            v8f acc = zero8();
#pragma unroll
            for (int kt = 0; kt < 4; ++kt) {
                v16h bf;
#pragma unroll
                for (int j = 0; j < 16; ++j) bf[j] = (_Float16)krow[kt * 32 + j];
                acc = __builtin_amdgcn_wmma_f32_16x16x32_f16(
                    false, qfrag[kt], false, bf, (short)0, acc, false, false);
            }
            // cross-wave reduction of the 16x32 score tile
#pragma unroll
            for (int i = 0; i < 8; ++i)
                atomicAdd(&S_lds[(rbase + i) * 32 + nt * 16 + n16], acc[i]);
        }

        // prefetch next K/V tiles (global_prefetch_b8)
        if (s0 + NTILE < SEQ) {
            __builtin_prefetch(Kb + (size_t)(s0 + NTILE + hi * 16 + n16) * DIM + chunk, 0, 1);
            __builtin_prefetch(Vb + (size_t)(s0 + NTILE) * DIM + chunk + lane * 4, 0, 1);
        }

        __syncthreads();   // all partial scores landed in S_lds

        // ---------- online softmax (wave 0; lane pair (L, L^16) shares row L%16) ----------
        if (wave == 0) {
            const int r  = n16;
            const int c0 = hi * 16;
            float vals[16];
            float mx = -__builtin_inff();
#pragma unroll
            for (int j = 0; j < 16; ++j) {
                vals[j] = S_lds[r * 32 + c0 + j] * 0.125f;  // scale = 1/sqrt(64)
                S_lds[r * 32 + c0 + j] = 0.0f;              // re-zero for next tile
                mx = fmaxf(mx, vals[j]);
            }
            mx = fmaxf(mx, __shfl_xor(mx, 16, 32));
            const float m_new = fmaxf(m_reg, mx);
            const float sc    = __expf(m_reg - m_new);
            float rs = 0.0f;
#pragma unroll
            for (int j = 0; j < 16; ++j) {
                float p = __expf(vals[j] - m_new);
                rs += p;
                P_lds[r * 32 + c0 + j] = (_Float16)p;
            }
            rs += __shfl_xor(rs, 16, 32);
            l_reg = l_reg * sc + rs;
            m_reg = m_new;
            if (hi == 0) scale_lds[r] = sc;
        }
        __syncthreads();   // P / scale ready

        // ---------- O = O*scale + P * V_chunk ----------
#pragma unroll
        for (int i = 0; i < 8; ++i) {
            const float sc = scale_lds[rbase + i];
#pragma unroll
            for (int ct = 0; ct < 8; ++ct) o[ct][i] *= sc;
        }

        // P as A-operand: two 16B vector loads from LDS
        v16h pf;
        {
            v8h lo = *(const v8h*)&P_lds[n16 * 32 + k0a];
            v8h hiv = *(const v8h*)&P_lds[n16 * 32 + k0a + 8];
#pragma unroll
            for (int j = 0; j < 8; ++j) { pf[j] = lo[j]; pf[j + 8] = hiv[j]; }
        }

        // V B-fragments: contiguous 32B reads from the transposed LDS tile
#pragma unroll
        for (int ct = 0; ct < 8; ++ct) {
            v16h bf = *(const v16h*)&vt[(ct * 16 + n16) * VTS + kb0];
            o[ct] = __builtin_amdgcn_wmma_f32_16x16x32_f16(
                false, pf, false, bf, (short)0, o[ct], false, false);
        }
    }

    // publish softmax denominators
    if (wave == 0 && hi == 0) l_lds[n16] = l_reg;
    __syncthreads();

    float rinv[8];
#pragma unroll
    for (int i = 0; i < 8; ++i) rinv[i] = 1.0f / l_lds[rbase + i];

    float* ob = out + (size_t)b * SEQ * DIM;
#pragma unroll
    for (int ct = 0; ct < 8; ++ct) {
#pragma unroll
        for (int i = 0; i < 8; ++i) {
            ob[(size_t)(q0 + rbase + i) * DIM + chunk + ct * 16 + n16] = o[ct][i] * rinv[i];
        }
    }
}

extern "C" void kernel_launch(void* const* d_in, const int* in_sizes, int n_in,
                              void* d_out, int out_size, void* d_ws, size_t ws_size,
                              hipStream_t stream) {
    (void)in_sizes; (void)n_in; (void)d_ws; (void)ws_size; (void)out_size;
    const float* q = (const float*)d_in[0];
    const float* k = (const float*)d_in[1];
    const float* v = (const float*)d_in[2];
    float* out = (float*)d_out;

    dim3 grid(4 * (SEQ / 16));   // batch * 16-row query strips = 1024 blocks
    dim3 block(256);             // 8 wave32 waves
    SparseAttention_flash_kernel<<<grid, block, 0, stream>>>(q, k, v, out);
}
